// StaticGraphEnocoder_65867618451714
// MI455X (gfx1250) — compile-verified, matrix operand order checked by reading, not run on hardware
//
#include <hip/hip_runtime.h>
#include <hip/hip_bf16.h>

typedef __bf16 bf16_t;
typedef __attribute__((ext_vector_type(16))) __bf16 v16bf;
typedef __attribute__((ext_vector_type(8)))  __bf16 v8bf;
typedef __attribute__((ext_vector_type(8)))  float  v8f;

static constexpr int Nn = 100000;   // nodes
static constexpr int Ed = 320000;   // edges per graph

__device__ __forceinline__ float wave_sum(float v) {
#pragma unroll
  for (int m = 16; m >= 1; m >>= 1) v += __shfl_xor(v, m, 32);
  return v;
}

// ---------------------------------------------------------------------------
// WMMA GEMM: C[M x (CT*16)] = A[M x K] (bf16, row-major) * Bt^T
// Bt is stored as [Ncols x K] bf16 (i.e. B pre-transposed) so that both A and
// B fragments are contiguous 16-byte loads per lane.
// One wave computes a 16-row stripe across all CT column tiles; the A
// fragments for the full K are loaded once into registers and reused.
// ---------------------------------------------------------------------------
template<int K, int CT>
__global__ __launch_bounds__(256) void gemm_bf16_wmma(
    const bf16_t* __restrict__ A, const bf16_t* __restrict__ Bt,
    float* __restrict__ C, int Mrows) {
  constexpr int KS = K / 32;
  constexpr int Ncols = CT * 16;
  const int wave = blockIdx.x * 8 + (threadIdx.x >> 5);
  if (wave * 16 >= Mrows) return;
  const int lane = threadIdx.x & 31;
  const int half = lane >> 4;       // which 16-lane half
  const int l16  = lane & 15;
  const int koff = half * 8;        // lanes 0-15: K 0..7 / 16..23 ; lanes 16-31: K 8..15 / 24..31

  const bf16_t* arow = A + (size_t)(wave * 16 + l16) * K;
  v16bf afrag[KS];
#pragma unroll
  for (int ks = 0; ks < KS; ++ks) {
    union { v16bf v; v8bf h[2]; } ua;
    ua.h[0] = *(const v8bf*)(arow + ks * 32 + koff);
    ua.h[1] = *(const v8bf*)(arow + ks * 32 + 16 + koff);
    afrag[ks] = ua.v;
  }

#pragma unroll
  for (int ct = 0; ct < CT; ++ct) {
    const bf16_t* bcol = Bt + (size_t)(ct * 16 + l16) * K;
    if (ct + 1 < CT)
      __builtin_prefetch(Bt + (size_t)((ct + 1) * 16 + l16) * K, 0, 1);
    v8f c = {0.f, 0.f, 0.f, 0.f, 0.f, 0.f, 0.f, 0.f};
#pragma unroll
    for (int ks = 0; ks < KS; ++ks) {
      union { v16bf v; v8bf h[2]; } ub;
      ub.h[0] = *(const v8bf*)(bcol + ks * 32 + koff);
      ub.h[1] = *(const v8bf*)(bcol + ks * 32 + 16 + koff);
      c = __builtin_amdgcn_wmma_f32_16x16x32_bf16(false, afrag[ks], false, ub.v,
                                                  (short)0, c, false, false);
    }
    // D layout: VGPR r -> row (r + 8*half), col l16
    float* crow = C + (size_t)(wave * 16 + half * 8) * Ncols + ct * 16 + l16;
#pragma unroll
    for (int r = 0; r < 8; ++r) crow[(size_t)r * Ncols] = c[r];
  }
}

// ---------------------------------------------------------------------------
// Prep kernels
// ---------------------------------------------------------------------------
__global__ __launch_bounds__(256) void f32_to_bf16_k(const float* __restrict__ in,
                                                     bf16_t* __restrict__ out, int n) {
  int i = blockIdx.x * 256 + threadIdx.x;
  if (i < n) out[i] = (bf16_t)in[i];
}

// W is [K x M] f32 row-major; Bt[n*K + k] = W[k*M + n] as bf16
__global__ __launch_bounds__(256) void transpose_bf16_k(const float* __restrict__ W,
                                                        bf16_t* __restrict__ Bt,
                                                        int K, int M) {
  int idx = blockIdx.x * 256 + threadIdx.x;
  if (idx >= K * M) return;
  int n = idx / K, k = idx - n * K;
  Bt[idx] = (bf16_t)W[k * M + n];
}

__global__ __launch_bounds__(256) void init_one_k(float* __restrict__ p, int n) {
  int i = blockIdx.x * 256 + threadIdx.x;
  if (i < n) p[i] = 1.0f;
}

__global__ __launch_bounds__(256) void count_k(const int* __restrict__ ids, int E,
                                               float* __restrict__ cnt) {
  int e = blockIdx.x * 256 + threadIdx.x;
  if (e < E) atomicAdd(&cnt[ids[e]], 1.0f);
}

__global__ __launch_bounds__(256) void rsqrt_k(float* __restrict__ dst,
                                               const float* __restrict__ src, int n) {
  int i = blockIdx.x * 256 + threadIdx.x;
  if (i < n) dst[i] = src[i] > 0.f ? rsqrtf(src[i]) : 0.f;
}

__global__ __launch_bounds__(256) void inv_inplace_k(float* __restrict__ p, int n) {
  int i = blockIdx.x * 256 + threadIdx.x;
  if (i < n) p[i] = p[i] > 0.f ? 1.0f / p[i] : 0.f;
}

// ---------------------------------------------------------------------------
// GCN aggregation: out[i] = h[i]*dis[i]^2 (self-loop) + sum_e h[src]*dis[src]*dis[dst]
// ---------------------------------------------------------------------------
template<int F>
__global__ __launch_bounds__(256) void gcn_self_k(const float* __restrict__ h,
                                                  const float* __restrict__ dis,
                                                  float* __restrict__ out, int rows) {
  int idx = blockIdx.x * 256 + threadIdx.x;
  if (idx >= rows * F) return;
  int i = idx / F;
  float d = dis[i];
  out[idx] = h[idx] * d * d;
}

template<int F>
__global__ __launch_bounds__(256) void gcn_scatter_k(const int* __restrict__ ei, int E,
                                                     const float* __restrict__ h,
                                                     const float* __restrict__ dis,
                                                     float* __restrict__ out) {
  constexpr int CH = F / 4;
  int t = blockIdx.x * 256 + threadIdx.x;
  int e = t / CH;
  if (e >= E) return;
  int f = (t - e * CH) * 4;
  int s = ei[e], d = ei[E + e];
  float nrm = dis[s] * dis[d];
  const float4 hv = *(const float4*)(h + (size_t)s * F + f);
  float* o = out + (size_t)d * F + f;
  atomicAdd(o + 0, hv.x * nrm);
  atomicAdd(o + 1, hv.y * nrm);
  atomicAdd(o + 2, hv.z * nrm);
  atomicAdd(o + 3, hv.w * nrm);
}

// ---------------------------------------------------------------------------
// Hypergraph segment scatter: out[dst_ids[e]] += src[src_ids[e]]
// ---------------------------------------------------------------------------
template<int F>
__global__ __launch_bounds__(256) void seg_scatter_k(const int* __restrict__ sid,
                                                     const int* __restrict__ did, int E,
                                                     const float* __restrict__ src,
                                                     float* __restrict__ out) {
  constexpr int CH = F / 4;
  int t = blockIdx.x * 256 + threadIdx.x;
  int e = t / CH;
  if (e >= E) return;
  int f = (t - e * CH) * 4;
  int s = sid[e], d = did[e];
  const float4 v = *(const float4*)(src + (size_t)s * F + f);
  float* o = out + (size_t)d * F + f;
  atomicAdd(o + 0, v.x);
  atomicAdd(o + 1, v.y);
  atomicAdd(o + 2, v.z);
  atomicAdd(o + 3, v.w);
}

template<int F>
__global__ __launch_bounds__(256) void row_scale_inv_k(float* __restrict__ x,
                                                       const float* __restrict__ cnt,
                                                       int rows) {
  int idx = blockIdx.x * 256 + threadIdx.x;
  if (idx >= rows * F) return;
  float c = cnt[idx / F];
  x[idx] *= (c > 0.f ? 1.0f / c : 0.f);
}

// ---------------------------------------------------------------------------
// Fused rowscale + bias + ReLU + LayerNorm. One wave per row.
// Writes f32 and/or bf16 output (whichever is non-null).
// ---------------------------------------------------------------------------
template<int F>
__global__ __launch_bounds__(256) void bias_relu_ln_k(
    const float* __restrict__ in, const float* __restrict__ rowscale,
    const float* __restrict__ bias, const float* __restrict__ gamma,
    const float* __restrict__ beta, float* __restrict__ out32,
    bf16_t* __restrict__ outbf, int rows) {
  constexpr int PER = F / 32;
  const int row = blockIdx.x * 8 + (threadIdx.x >> 5);
  if (row >= rows) return;
  const int lane = threadIdx.x & 31;
  const float s = rowscale ? rowscale[row] : 1.0f;
  const size_t base = (size_t)row * F + lane * PER;
  float v[PER];
  float sum = 0.f;
#pragma unroll
  for (int i = 0; i < PER; ++i) {
    float t = in[base + i] * s + bias[lane * PER + i];
    v[i] = t > 0.f ? t : 0.f;
    sum += v[i];
  }
  const float mean = wave_sum(sum) * (1.0f / F);
  float s2 = 0.f;
#pragma unroll
  for (int i = 0; i < PER; ++i) { float d = v[i] - mean; s2 += d * d; }
  const float rstd = rsqrtf(wave_sum(s2) * (1.0f / F) + 1e-5f);
#pragma unroll
  for (int i = 0; i < PER; ++i) {
    float o = (v[i] - mean) * rstd * gamma[lane * PER + i] + beta[lane * PER + i];
    if (out32) out32[base + i] = o;
    if (outbf) outbf[base + i] = (bf16_t)o;
  }
}

// ---------------------------------------------------------------------------
// Per-node 3-key attention + final LayerNorm. One wave per node (H=128).
// ---------------------------------------------------------------------------
__global__ __launch_bounds__(256) void fuse_attn_ln_k(
    const float* __restrict__ q, const float* __restrict__ x1,
    const float* __restrict__ x2, const float* __restrict__ x3,
    const float* __restrict__ gamma, const float* __restrict__ beta,
    float* __restrict__ out, int rows) {
  const int row = blockIdx.x * 8 + (threadIdx.x >> 5);
  if (row >= rows) return;
  const int lane = threadIdx.x & 31;
  const size_t base = (size_t)row * 128 + lane * 4;
  const float4 qv = *(const float4*)(q  + base);
  const float4 a  = *(const float4*)(x1 + base);
  const float4 b  = *(const float4*)(x2 + base);
  const float4 c  = *(const float4*)(x3 + base);
  float d1 = qv.x * a.x + qv.y * a.y + qv.z * a.z + qv.w * a.w;
  float d2 = qv.x * b.x + qv.y * b.y + qv.z * b.z + qv.w * b.w;
  float d3 = qv.x * c.x + qv.y * c.y + qv.z * c.z + qv.w * c.w;
  const float sc = 0.08838834764831845f;  // 1/sqrt(128)
  d1 = wave_sum(d1) * sc;
  d2 = wave_sum(d2) * sc;
  d3 = wave_sum(d3) * sc;
  float mx = fmaxf(d1, fmaxf(d2, d3));
  float e1 = __expf(d1 - mx), e2 = __expf(d2 - mx), e3 = __expf(d3 - mx);
  float inv = 1.0f / (e1 + e2 + e3);
  float w1 = e1 * inv, w2 = e2 * inv, w3 = e3 * inv;
  float o[4] = { w1 * a.x + w2 * b.x + w3 * c.x,
                 w1 * a.y + w2 * b.y + w3 * c.y,
                 w1 * a.z + w2 * b.z + w3 * c.z,
                 w1 * a.w + w2 * b.w + w3 * c.w };
  float sum = o[0] + o[1] + o[2] + o[3];
  const float mean = wave_sum(sum) * (1.0f / 128.0f);
  float s2 = 0.f;
#pragma unroll
  for (int i = 0; i < 4; ++i) { float d = o[i] - mean; s2 += d * d; }
  const float rstd = rsqrtf(wave_sum(s2) * (1.0f / 128.0f) + 1e-5f);
#pragma unroll
  for (int i = 0; i < 4; ++i)
    out[base + i] = (o[i] - mean) * rstd * gamma[lane * 4 + i] + beta[lane * 4 + i];
}

// ---------------------------------------------------------------------------
// Host orchestration
// ---------------------------------------------------------------------------
static inline unsigned gblk(long n) { return (unsigned)((n + 255) / 256); }
static constexpr unsigned GEMM_GRID = (Nn / 16 + 7) / 8;  // 6250 waves, 8 per block

static void run_gcn(const bf16_t* embBF, const int* ei,
                    const float* W1, const float* b1, const float* g1, const float* be1,
                    const float* W2, const float* b2, const float* g2, const float* be2,
                    bf16_t* w1t, bf16_t* w2t, float* hbuf, float* agg, bf16_t* lnbf,
                    float* deg, float* dis, float* xout, hipStream_t s) {
  init_one_k<<<gblk(Nn), 256, 0, s>>>(deg, Nn);                       // self-loop deg = 1
  count_k<<<gblk(Ed), 256, 0, s>>>(ei + Ed, Ed, deg);                 // + in-degree
  rsqrt_k<<<gblk(Nn), 256, 0, s>>>(dis, deg, Nn);
  // layer 1: [N,128] x [128,256]
  transpose_bf16_k<<<gblk(128 * 256), 256, 0, s>>>(W1, w1t, 128, 256);
  gemm_bf16_wmma<128, 16><<<GEMM_GRID, 256, 0, s>>>(embBF, w1t, hbuf, Nn);
  gcn_self_k<256><<<gblk((long)Nn * 256), 256, 0, s>>>(hbuf, dis, agg, Nn);
  gcn_scatter_k<256><<<gblk((long)Ed * 64), 256, 0, s>>>(ei, Ed, hbuf, dis, agg);
  bias_relu_ln_k<256><<<(Nn + 7) / 8, 256, 0, s>>>(agg, nullptr, b1, g1, be1,
                                                   nullptr, lnbf, Nn);
  // layer 2: [N,256] x [256,128]
  transpose_bf16_k<<<gblk(256 * 128), 256, 0, s>>>(W2, w2t, 256, 128);
  gemm_bf16_wmma<256, 8><<<GEMM_GRID, 256, 0, s>>>(lnbf, w2t, hbuf, Nn);
  gcn_self_k<128><<<gblk((long)Nn * 128), 256, 0, s>>>(hbuf, dis, agg, Nn);
  gcn_scatter_k<128><<<gblk((long)Ed * 32), 256, 0, s>>>(ei, Ed, hbuf, dis, agg);
  bias_relu_ln_k<128><<<(Nn + 7) / 8, 256, 0, s>>>(agg, nullptr, b2, g2, be2,
                                                   xout, nullptr, Nn);
}

static void run_hgnn(const bf16_t* embBF, const int* ei,
                     const float* W1, const float* b1, const float* g1, const float* be1,
                     const float* W2, const float* b2, const float* g2, const float* be2,
                     bf16_t* w1t, bf16_t* w2t, float* hbuf, float* agg, bf16_t* lnbf,
                     float* hef, float* cntD, float* cntB, float* xout, hipStream_t s) {
  hipMemsetAsync(cntD, 0, (size_t)Nn * 4, s);
  hipMemsetAsync(cntB, 0, (size_t)Nn * 4, s);
  count_k<<<gblk(Ed), 256, 0, s>>>(ei, Ed, cntD);        // node degree D
  count_k<<<gblk(Ed), 256, 0, s>>>(ei + Ed, Ed, cntB);   // hyperedge degree B
  inv_inplace_k<<<gblk(Nn), 256, 0, s>>>(cntD, Nn);      // cntD := D^-1
  // layer 1
  transpose_bf16_k<<<gblk(128 * 256), 256, 0, s>>>(W1, w1t, 128, 256);
  gemm_bf16_wmma<128, 16><<<GEMM_GRID, 256, 0, s>>>(embBF, w1t, hbuf, Nn);
  hipMemsetAsync(hef, 0, (size_t)Nn * 256 * 4, s);
  seg_scatter_k<256><<<gblk((long)Ed * 64), 256, 0, s>>>(ei, ei + Ed, Ed, hbuf, hef);
  row_scale_inv_k<256><<<gblk((long)Nn * 256), 256, 0, s>>>(hef, cntB, Nn);
  hipMemsetAsync(agg, 0, (size_t)Nn * 256 * 4, s);
  seg_scatter_k<256><<<gblk((long)Ed * 64), 256, 0, s>>>(ei + Ed, ei, Ed, hef, agg);
  bias_relu_ln_k<256><<<(Nn + 7) / 8, 256, 0, s>>>(agg, cntD, b1, g1, be1,
                                                   nullptr, lnbf, Nn);
  // layer 2
  transpose_bf16_k<<<gblk(256 * 128), 256, 0, s>>>(W2, w2t, 256, 128);
  gemm_bf16_wmma<256, 8><<<GEMM_GRID, 256, 0, s>>>(lnbf, w2t, hbuf, Nn);
  hipMemsetAsync(hef, 0, (size_t)Nn * 128 * 4, s);
  seg_scatter_k<128><<<gblk((long)Ed * 32), 256, 0, s>>>(ei, ei + Ed, Ed, hbuf, hef);
  row_scale_inv_k<128><<<gblk((long)Nn * 128), 256, 0, s>>>(hef, cntB, Nn);
  hipMemsetAsync(agg, 0, (size_t)Nn * 128 * 4, s);
  seg_scatter_k<128><<<gblk((long)Ed * 32), 256, 0, s>>>(ei + Ed, ei, Ed, hef, agg);
  bias_relu_ln_k<128><<<(Nn + 7) / 8, 256, 0, s>>>(agg, cntD, b2, g2, be2,
                                                   xout, nullptr, Nn);
}

extern "C" void kernel_launch(void* const* d_in, const int* in_sizes, int n_in,
                              void* d_out, int out_size, void* d_ws, size_t ws_size,
                              hipStream_t stream) {
  (void)in_sizes; (void)n_in; (void)out_size; (void)ws_size;
  const float* emb    = (const float*)d_in[0];
  const int*   e_soc  = (const int*)d_in[1];
  const int*   e_dif  = (const int*)d_in[2];
  const int*   e_hyp  = (const int*)d_in[3];
  const float* fuse_g = (const float*)d_in[4];
  const float* fuse_b = (const float*)d_in[5];
  // per-branch params: W1,b1,g1,be1,W2,b2,g2,be2 at 6+8*branch
  const float* P[3][8];
  for (int br = 0; br < 3; ++br)
    for (int j = 0; j < 8; ++j) P[br][j] = (const float*)d_in[6 + br * 8 + j];

  // workspace carve-out
  char* ws = (char*)d_ws;
  size_t off = 0;
  auto carve = [&](size_t bytes) -> void* {
    void* p = ws + off;
    off += (bytes + 255) & ~(size_t)255;
    return p;
  };
  bf16_t* embBF = (bf16_t*)carve((size_t)Nn * 128 * 2);
  bf16_t* w1t   = (bf16_t*)carve((size_t)256 * 128 * 2);
  bf16_t* w2t   = (bf16_t*)carve((size_t)128 * 256 * 2);
  float*  hbuf  = (float*)carve((size_t)Nn * 256 * 4);
  float*  agg   = (float*)carve((size_t)Nn * 256 * 4);
  bf16_t* lnbf  = (bf16_t*)carve((size_t)Nn * 256 * 2);
  float*  hef   = (float*)carve((size_t)Nn * 256 * 4);
  float*  cntA  = (float*)carve((size_t)Nn * 4);  // deg / D^-1
  float*  cntB  = (float*)carve((size_t)Nn * 4);  // dis / B
  float*  x1    = (float*)carve((size_t)Nn * 128 * 4);
  float*  x2    = (float*)carve((size_t)Nn * 128 * 4);
  float*  x3    = (float*)carve((size_t)Nn * 128 * 4);

  f32_to_bf16_k<<<gblk((long)Nn * 128), 256, 0, stream>>>(emb, embBF, Nn * 128);

  run_gcn(embBF, e_soc, P[0][0], P[0][1], P[0][2], P[0][3], P[0][4], P[0][5], P[0][6], P[0][7],
          w1t, w2t, hbuf, agg, lnbf, cntA, cntB, x1, stream);
  run_gcn(embBF, e_dif, P[1][0], P[1][1], P[1][2], P[1][3], P[1][4], P[1][5], P[1][6], P[1][7],
          w1t, w2t, hbuf, agg, lnbf, cntA, cntB, x2, stream);
  run_hgnn(embBF, e_hyp, P[2][0], P[2][1], P[2][2], P[2][3], P[2][4], P[2][5], P[2][6], P[2][7],
           w1t, w2t, hbuf, agg, lnbf, hef, cntA, cntB, x3, stream);

  fuse_attn_ln_k<<<(Nn + 7) / 8, 256, 0, stream>>>(emb, x1, x2, x3, fuse_g, fuse_b,
                                                   (float*)d_out, Nn);
}
